// Renderer_27401891349284
// MI455X (gfx1250) — compile-verified, matrix operand order checked by reading, not run on hardware
//
#include <hip/hip_runtime.h>
#include <hip/hip_bf16.h>
#include <math.h>

typedef _Float16 h16;
typedef __attribute__((ext_vector_type(16))) _Float16 v16h;
typedef __attribute__((ext_vector_type(8)))  float    v8f;

#define HH 512
#define WWD 512
#define HWPIX (HH * WWD)
#define NPTS 100000
#define CPAD 40  // LDS per-pixel channel stride (halfs): 80B, 16B-aligned, 2-way banked

// ---------------------------------------------------------------------------
// WMMA fragment layout (gfx1250, wave32)
//
// A (16x32 f16, MxK):  lane<16: M=lane, K pairs {0..7,16..23};
//                      lane>=16: M=lane-16, K pairs {8..15,24..31}
//   k(v,h) = (v<4?0:16) + (lane>=16?8:0) + (v&3)*2 + h
// B (32x16 f16, KxN):  mirror with N = lane&15, same K pattern.
// C/D (16x16 f32):     lane<16: N=lane, vgpr r -> M=r;
//                      lane>=16: N=lane-16, vgpr r -> M=r+8
//
// Weights are pre-packed so each lane's B fragment is 32 contiguous bytes:
//   packed u32 index = ((kt*NT + nt)*32 + lane)*8 + v    (u32 = f16 pair)
// Activations are NHWC f16 (channel-innermost) so async global->LDS copies
// land directly in the channel-innermost layout A-fragments want.
// ---------------------------------------------------------------------------

__device__ __forceinline__ void load_bfrag(const h16* __restrict__ Wp,
                                           int kt, int nt, int NT, int lane,
                                           v16h& b) {
  const uint4* q = (const uint4*)(Wp + (size_t)((kt * NT + nt) * 32 + lane) * 16);
  ((uint4*)&b)[0] = q[0];
  ((uint4*)&b)[1] = q[1];
}

// ---------------------------------------------------------------------------
// Weight packing: fp32 -> f16 in B-fragment order
// ---------------------------------------------------------------------------

// Dense [K x N] row-major, K padded to Kpad (zero fill)
__global__ void pack_dense(const float* __restrict__ src, h16* __restrict__ dst,
                           int K, int Kpad, int N) {
  const int NT = N >> 4;
  const int KT = Kpad >> 5;
  const int total = KT * NT * 256; // u32 count
  int i = blockIdx.x * 256 + threadIdx.x;
  if (i >= total) return;
  int v = i & 7;
  int lane = (i >> 3) & 31;
  int rest = i >> 8;
  int nt = rest % NT;
  int kt = rest / NT;
  int k = kt * 32 + ((v < 4) ? 0 : 16) + ((lane >= 16) ? 8 : 0) + (v & 3) * 2;
  int n = nt * 16 + (lane & 15);
  h16 lo = (k < K)     ? (h16)src[(size_t)k * N + n]       : (h16)0.f;
  h16 hi = (k + 1 < K) ? (h16)src[(size_t)(k + 1) * N + n] : (h16)0.f;
  dst[2 * i + 0] = lo;
  dst[2 * i + 1] = hi;
}

// Conv OIHW 3x3 -> implicit-GEMM B matrix: K = tap*Ipad + ci, N = o
__global__ void pack_conv(const float* __restrict__ src, h16* __restrict__ dst,
                          int O, int I, int Ipad) {
  const int NT = O >> 4;
  const int KT = (9 * Ipad) >> 5;
  const int total = KT * NT * 256;
  int i = blockIdx.x * 256 + threadIdx.x;
  if (i >= total) return;
  int v = i & 7;
  int lane = (i >> 3) & 31;
  int rest = i >> 8;
  int nt = rest % NT;
  int kt = rest / NT;
  int k = kt * 32 + ((v < 4) ? 0 : 16) + ((lane >= 16) ? 8 : 0) + (v & 3) * 2;
  int o = nt * 16 + (lane & 15);
  int tap = k / Ipad;       // ky*3+kx
  int ci = k - tap * Ipad;  // even; ci+1 stays in same tap (Ipad even)
  h16 lo = (ci < I)     ? (h16)src[((size_t)o * I + ci) * 9 + tap]     : (h16)0.f;
  h16 hi = (ci + 1 < I) ? (h16)src[((size_t)o * I + ci + 1) * 9 + tap] : (h16)0.f;
  dst[2 * i + 0] = lo;
  dst[2 * i + 1] = hi;
}

// ---------------------------------------------------------------------------
// MLP: one wave owns 16 pixels; ping-pong 16x160 f16 tiles in LDS.
// ---------------------------------------------------------------------------

__device__ __forceinline__ void mlp_layer(const h16* __restrict__ actIn,
                                          h16* __restrict__ actOut,
                                          const h16* __restrict__ Wp,
                                          const float* __restrict__ bias,
                                          int ktiles, int ntiles, int lane) {
  const int khi = (lane >= 16) ? 8 : 0;
  const int m = lane & 15;
  const int n = lane & 15;
  const int mhi = (lane >= 16) ? 8 : 0;
  for (int nt = 0; nt < ntiles; ++nt) {
    v8f acc = {};
    for (int kt = 0; kt < ktiles; ++kt) {
      v16h a, b;
      unsigned int* au = (unsigned int*)&a;
#pragma unroll
      for (int v = 0; v < 8; ++v) {
        int k = kt * 32 + ((v < 4) ? 0 : 16) + khi + (v & 3) * 2;
        au[v] = *(const unsigned int*)(actIn + m * 160 + k);
      }
      load_bfrag(Wp, kt, nt, ntiles, lane, b);
      acc = __builtin_amdgcn_wmma_f32_16x16x32_f16(false, a, false, b,
                                                   (short)0, acc, false, false);
    }
    float bv = bias[nt * 16 + n];
#pragma unroll
    for (int r = 0; r < 8; ++r) {
      float vv = acc[r] + bv;
      vv = vv > 0.f ? vv : 0.f;
      actOut[(r + mhi) * 160 + nt * 16 + n] = (h16)vv;
    }
  }
}

__global__ void __launch_bounds__(256)
mlp_kernel(const int* __restrict__ zbuf, const float* __restrict__ ray,
           const float* __restrict__ xyz_o,
           const h16* __restrict__ w0p, const float* __restrict__ b0,
           const h16* __restrict__ w1p, const float* __restrict__ b1,
           const h16* __restrict__ w2p, const float* __restrict__ b2,
           const h16* __restrict__ w3p, const float* __restrict__ b3,
           const h16* __restrict__ wdp, const float* __restrict__ bd,
           const h16* __restrict__ wop, const float* __restrict__ bo,
           h16* __restrict__ fmap, float* __restrict__ fea) {
  extern __shared__ h16 smem[];
  const int tid = threadIdx.x;
  const int wave = tid >> 5;
  const int lane = tid & 31;
  h16* bufA = smem + wave * (2 * 16 * 160);
  h16* bufB = bufA + 16 * 160;
  const int p0 = blockIdx.x * 128 + wave * 16;

  // positional encodings (lanes 0..15, one pixel each)
  if (lane < 16) {
    const int p = p0 + lane;
    int zb = zbuf[p];
    int idx = zb < 0 ? 0 : (zb > NPTS - 1 ? NPTS - 1 : zb);
    float px = xyz_o[idx * 3 + 0];
    float py = xyz_o[idx * 3 + 1];
    float pz = xyz_o[idx * 3 + 2];
    h16* row = bufA + lane * 160;
    row[0] = (h16)px; row[1] = (h16)py; row[2] = (h16)pz;
    float f = 1.0f;
#pragma unroll
    for (int l = 0; l < 10; ++l) {
      float s, c;
      __sincosf(px * f, &s, &c); row[3 + 6 * l + 0] = (h16)s; row[3 + 6 * l + 3] = (h16)c;
      __sincosf(py * f, &s, &c); row[3 + 6 * l + 1] = (h16)s; row[3 + 6 * l + 4] = (h16)c;
      __sincosf(pz * f, &s, &c); row[3 + 6 * l + 2] = (h16)s; row[3 + 6 * l + 5] = (h16)c;
      f *= 2.0f;
    }
    row[63] = (h16)0.f;
    float dx = ray[p * 7 + 3], dy = ray[p * 7 + 4], dz = ray[p * 7 + 5];
    row[128] = (h16)dx; row[129] = (h16)dy; row[130] = (h16)dz;
    f = 1.0f;
#pragma unroll
    for (int l = 0; l < 4; ++l) {
      float s, c;
      __sincosf(dx * f, &s, &c); row[131 + 6 * l + 0] = (h16)s; row[131 + 6 * l + 3] = (h16)c;
      __sincosf(dy * f, &s, &c); row[131 + 6 * l + 1] = (h16)s; row[131 + 6 * l + 4] = (h16)c;
      __sincosf(dz * f, &s, &c); row[131 + 6 * l + 2] = (h16)s; row[131 + 6 * l + 5] = (h16)c;
      f *= 2.0f;
    }
#pragma unroll
    for (int k = 155; k < 160; ++k) row[k] = (h16)0.f;
  }
  __syncthreads();
  mlp_layer(bufA, bufB, w0p, b0, 2, 8, lane);  // 63(64) -> 128
  __syncthreads();
  mlp_layer(bufB, bufA, w1p, b1, 4, 8, lane);  // 128 -> 128
  __syncthreads();
  mlp_layer(bufA, bufB, w2p, b2, 4, 8, lane);
  __syncthreads();
  mlp_layer(bufB, bufA, w3p, b3, 4, 8, lane);  // bufA[.,128..159] still holds dir enc
  __syncthreads();
  mlp_layer(bufA, bufB, wdp, bd, 5, 8, lane);  // 155(160) -> 128
  __syncthreads();

  // output layer 128 -> 16, no relu, masked stores (NHWC, 32-ch padded)
  {
    const int khi = (lane >= 16) ? 8 : 0;
    const int m = lane & 15;
    v8f acc = {};
    for (int kt = 0; kt < 4; ++kt) {
      v16h a, b;
      unsigned int* au = (unsigned int*)&a;
#pragma unroll
      for (int v = 0; v < 8; ++v) {
        int k = kt * 32 + ((v < 4) ? 0 : 16) + khi + (v & 3) * 2;
        au[v] = *(const unsigned int*)(bufB + m * 160 + k);
      }
      load_bfrag(wop, kt, 0, 1, lane, b);
      acc = __builtin_amdgcn_wmma_f32_16x16x32_f16(false, a, false, b,
                                                   (short)0, acc, false, false);
    }
    const int n = lane & 15;
    const int mhi = (lane >= 16) ? 8 : 0;
    float bv = bo[n];
#pragma unroll
    for (int r = 0; r < 8; ++r) {
      int p = p0 + r + mhi;
      float vv = acc[r] + bv;
      if (zbuf[p] <= 0) vv = 0.f;
      fmap[(size_t)p * 32 + n] = (h16)vv;        // NHWC f16, channel n
      fmap[(size_t)p * 32 + 16 + n] = (h16)0.f;  // zero pad channels 16..31
      if (n < 3) fea[(size_t)p * 3 + n] = vv;    // fea_map output (f32)
    }
  }
}

// ---------------------------------------------------------------------------
// Implicit-GEMM 3x3 SAME conv + bias + relu, NHWC activations.
// Block = 256 threads = 8 waves = 128 consecutive pixels in one row.
// Per 32-channel chunk: async-copy (GLOBAL_LOAD_ASYNC_TO_LDS_B128, ASYNCcnt)
// a 3 x 132-pixel x 32-channel halo tile into LDS, double-buffered so the
// next chunk streams in while WMMAs consume the current one. Borders are
// zero-filled with plain ds stores.
// ---------------------------------------------------------------------------

__device__ __forceinline__ void stage_tile_async(const h16* __restrict__ in, int Cin,
                                                 int H, int W, int y, int x0, int kc,
                                                 h16* __restrict__ buf, unsigned bufoff,
                                                 int tid) {
  for (int t = tid; t < 3 * 132 * 4; t += 256) {
    int q = t & 3;          // 16B sub-chunk (8 channels)
    int pc = t >> 2;        // 0..395 pixel slot
    int col = pc % 132;
    int r = pc / 132;
    int yy = y + r - 1;
    int xx = x0 + col - 1;
    int idx = pc * CPAD + q * 8;  // halfs
    if (yy >= 0 && yy < H && xx >= 0 && xx < W) {
      unsigned lds_off = bufoff + (unsigned)(idx * 2);
      const h16* gp = in + ((size_t)yy * W + xx) * Cin + (kc * 32 + q * 8);
      unsigned long long ga = (unsigned long long)(size_t)gp;
      asm volatile("global_load_async_to_lds_b128 %0, %1, off"
                   :: "v"(lds_off), "v"(ga) : "memory");
    } else {
      uint4 z = make_uint4(0u, 0u, 0u, 0u);
      *(uint4*)(buf + idx) = z;  // zero-fill border
    }
  }
}

template <int NT>
__global__ void __launch_bounds__(256)
conv3x3_kernel(const h16* __restrict__ in, const h16* __restrict__ Wp,
               const float* __restrict__ bias, h16* __restrict__ out,
               int Cin, int H, int W) {
  __shared__ h16 tile[2][3 * 132 * CPAD];
  const int tid = threadIdx.x;
  const int wave = tid >> 5;
  const int lane = tid & 31;
  const int pblk = blockIdx.x * 128;
  const int y = pblk / W;
  const int x0 = pblk - y * W;
  const int khi = (lane >= 16) ? 8 : 0;
  const int m = lane & 15;
  const int colw = wave * 16 + m;

  // wave-relative LDS byte offsets of the two buffers (generic-pointer low bits)
  const unsigned base0 = (unsigned)(unsigned long long)(void*)&tile[0][0];
  const unsigned base1 = (unsigned)(unsigned long long)(void*)&tile[1][0];

  v8f zero = {};
  v8f acc[NT];
#pragma unroll
  for (int i = 0; i < NT; ++i) acc[i] = zero;

  const int kchunks = Cin >> 5;

  stage_tile_async(in, Cin, H, W, y, x0, 0, &tile[0][0], base0, tid);

  for (int kc = 0; kc < kchunks; ++kc) {
    asm volatile("s_wait_asynccnt 0x0" ::: "memory");  // own async writes landed
    __syncthreads();                                   // everyone's staging done
    if (kc + 1 < kchunks)
      stage_tile_async(in, Cin, H, W, y, x0, kc + 1, &tile[(kc + 1) & 1][0],
                       ((kc + 1) & 1) ? base1 : base0, tid);
    const h16* buf = &tile[kc & 1][0];
#pragma unroll
    for (int tap = 0; tap < 9; ++tap) {
      const int dy = tap / 3;
      const int dx = tap - dy * 3;
      v16h a;
      unsigned int* au = (unsigned int*)&a;
      const int abase = (dy * 132 + colw + dx) * CPAD;
#pragma unroll
      for (int v = 0; v < 8; ++v) {
        int k = ((v < 4) ? 0 : 16) + khi + (v & 3) * 2;
        au[v] = *(const unsigned int*)(buf + abase + k);
      }
      const int ktw = tap * kchunks + kc;
#pragma unroll
      for (int nt = 0; nt < NT; ++nt) {
        v16h b;
        const uint4* q = (const uint4*)(Wp + (size_t)((ktw * NT + nt) * 32 + lane) * 16);
        ((uint4*)&b)[0] = q[0];
        ((uint4*)&b)[1] = q[1];
        acc[nt] = __builtin_amdgcn_wmma_f32_16x16x32_f16(false, a, false, b,
                                                         (short)0, acc[nt], false, false);
      }
    }
  }

  const int Cout = NT * 16;
  const int n = lane & 15;
  const int mhi = (lane >= 16) ? 8 : 0;
  const int xw = x0 + wave * 16;
#pragma unroll
  for (int nt = 0; nt < NT; ++nt) {
    float bv = bias[nt * 16 + n];
#pragma unroll
    for (int r = 0; r < 8; ++r) {
      float vv = acc[nt][r] + bv;
      vv = vv > 0.f ? vv : 0.f;
      out[((size_t)y * W + (xw + r + mhi)) * Cout + nt * 16 + n] = (h16)vv;
    }
  }
}

// ---------------------------------------------------------------------------
// Elementwise glue (NHWC)
// ---------------------------------------------------------------------------

__global__ void pool2_kernel(const h16* __restrict__ in, h16* __restrict__ out,
                             int C, int H, int W) {
  int Ho = H >> 1, Wo = W >> 1;
  size_t total = (size_t)Ho * Wo * C;
  size_t i = (size_t)blockIdx.x * 256 + threadIdx.x;
  if (i >= total) return;
  int c = (int)(i % C);
  size_t pix = i / C;
  int xo = (int)(pix % Wo);
  int yo = (int)(pix / Wo);
  const h16* p = in + (((size_t)(yo * 2) * W) + xo * 2) * C + c;
  float a = (float)p[0], b = (float)p[C], cc = (float)p[(size_t)W * C],
        d = (float)p[(size_t)W * C + C];
  out[i] = (h16)fmaxf(fmaxf(a, b), fmaxf(cc, d));
}

__global__ void upcat_kernel(const h16* __restrict__ up, const h16* __restrict__ skip,
                             h16* __restrict__ out, int C1, int C2, int Ho, int Wo) {
  int Ct = C1 + C2;
  size_t total = (size_t)Ho * Wo * Ct;
  size_t i = (size_t)blockIdx.x * 256 + threadIdx.x;
  if (i >= total) return;
  int c = (int)(i % Ct);
  size_t pix = i / Ct;
  int x = (int)(pix % Wo);
  int yy = (int)(pix / Wo);
  h16 vv;
  if (c < C1) {
    int Wi = Wo >> 1;
    vv = up[(((size_t)(yy >> 1) * Wi) + (x >> 1)) * C1 + c];
  } else {
    vv = skip[pix * C2 + (c - C1)];
  }
  out[i] = vv;
}

__global__ void final_kernel(const h16* __restrict__ u1, const float* __restrict__ wout,
                             const float* __restrict__ bout, const int* __restrict__ zbuf,
                             const float* __restrict__ gt, float* __restrict__ dst) {
  int p = blockIdx.x * 256 + threadIdx.x;
  if (p >= HWPIX) return;
  float a0 = bout[0], a1 = bout[1], a2 = bout[2];
  const h16* up = u1 + (size_t)p * 32;  // NHWC: contiguous 32 channels
#pragma unroll 4
  for (int c = 0; c < 32; ++c) {
    float vv = (float)up[c];
    a0 += vv * wout[c];
    a1 += vv * wout[32 + c];
    a2 += vv * wout[64 + c];
  }
  float pm = (zbuf[p] > 0) ? 1.f : 0.f;
  float s0 = 1.f / (1.f + __expf(-a0));
  float s1 = 1.f / (1.f + __expf(-a1));
  float s2 = 1.f / (1.f + __expf(-a2));
  dst[(size_t)p * 3 + 0] = s0 * pm + (1.f - pm);
  dst[(size_t)p * 3 + 1] = s1 * pm + (1.f - pm);
  dst[(size_t)p * 3 + 2] = s2 * pm + (1.f - pm);
  // gt passthrough (second tuple element)
  dst[(size_t)HWPIX * 3 + (size_t)p * 3 + 0] = gt[p * 3 + 0];
  dst[(size_t)HWPIX * 3 + (size_t)p * 3 + 1] = gt[p * 3 + 1];
  dst[(size_t)HWPIX * 3 + (size_t)p * 3 + 2] = gt[p * 3 + 2];
}

// ---------------------------------------------------------------------------
// Host launcher
// ---------------------------------------------------------------------------

extern "C" void kernel_launch(void* const* d_in, const int* in_sizes, int n_in,
                              void* d_out, int out_size, void* d_ws, size_t ws_size,
                              hipStream_t stream) {
  (void)in_sizes; (void)n_in; (void)out_size; (void)ws_size;

  const int*   zbuf = (const int*)d_in[0];
  const float* ray  = (const float*)d_in[1];
  const float* gt   = (const float*)d_in[2];
  const float* xyz  = (const float*)d_in[3];
  // d_in[4] = isTrain (unused)
  const float* W0 = (const float*)d_in[5];  const float* b0 = (const float*)d_in[6];
  const float* W1 = (const float*)d_in[7];  const float* b1 = (const float*)d_in[8];
  const float* W2 = (const float*)d_in[9];  const float* b2 = (const float*)d_in[10];
  const float* W3 = (const float*)d_in[11]; const float* b3 = (const float*)d_in[12];
  const float* Wd = (const float*)d_in[13]; const float* bd = (const float*)d_in[14];
  const float* Wo = (const float*)d_in[15]; const float* bo = (const float*)d_in[16];
  const float* w1a = (const float*)d_in[17]; const float* b1a = (const float*)d_in[18];
  const float* w1b = (const float*)d_in[19]; const float* b1b = (const float*)d_in[20];
  const float* w2a = (const float*)d_in[21]; const float* b2a = (const float*)d_in[22];
  const float* w2b = (const float*)d_in[23]; const float* b2b = (const float*)d_in[24];
  const float* w3a = (const float*)d_in[25]; const float* b3a = (const float*)d_in[26];
  const float* w3b = (const float*)d_in[27]; const float* b3b = (const float*)d_in[28];
  const float* w4a = (const float*)d_in[29]; const float* b4a = (const float*)d_in[30];
  const float* w4b = (const float*)d_in[31]; const float* b4b = (const float*)d_in[32];
  const float* w5a = (const float*)d_in[33]; const float* b5a = (const float*)d_in[34];
  const float* w5b = (const float*)d_in[35]; const float* b5b = (const float*)d_in[36];
  const float* wout = (const float*)d_in[37]; const float* bout = (const float*)d_in[38];

  float* out = (float*)d_out;
  h16* ws = (h16*)d_ws;

  // ---- workspace arena (f16 elements), lifetime-based reuse ----
  const size_t HW = (size_t)HWPIX;
  h16* A = ws;              // 32*HW
  h16* B = ws + 32 * HW;    // 32*HW
  h16* C = ws + 64 * HW;    // 32*HW
  h16* D = ws + 96 * HW;    // 96*HW
  h16* WGT = ws + 192 * HW;

  h16* FMAP = A;            // 512^2 x 32 ch (16 real + 16 zero pad), NHWC
  h16* T1 = B;              // 512^2 x 32
  h16* E1 = C;              // 512^2 x 32 (skip, long-lived)
  h16* P1 = B;              // 256^2 x 32
  h16* T2 = B + 8 * HW;     // 256^2 x 64
  h16* E2 = A;              // 256^2 x 64 (skip)
  h16* P2 = A + 16 * HW;    // 128^2 x 64
  h16* T3 = B;              // 128^2 x 128
  h16* BMu = B + 8 * HW;    // 128^2 x 128
  h16* CAT2 = D;            // 256^2 x 192
  h16* T4 = B;              // 256^2 x 64
  h16* U2 = A;              // 256^2 x 64
  h16* CAT1 = D;            // 512^2 x 96
  h16* T5 = A;              // 512^2 x 32
  h16* U1 = B;              // 512^2 x 32

  // packed weights
  h16* p_ = WGT;
  h16* w0p = p_; p_ += 64 * 128;
  h16* w1p = p_; p_ += 128 * 128;
  h16* w2p = p_; p_ += 128 * 128;
  h16* w3p = p_; p_ += 128 * 128;
  h16* wdp = p_; p_ += 160 * 128;
  h16* wop = p_; p_ += 128 * 16;
  h16* c1a = p_; p_ += 9 * 32 * 32;
  h16* c1b = p_; p_ += 9 * 32 * 32;
  h16* c2a = p_; p_ += 9 * 32 * 64;
  h16* c2b = p_; p_ += 9 * 64 * 64;
  h16* c3a = p_; p_ += 9 * 64 * 128;
  h16* c3b = p_; p_ += 9 * 128 * 128;
  h16* c4a = p_; p_ += 9 * 192 * 64;
  h16* c4b = p_; p_ += 9 * 64 * 64;
  h16* c5a = p_; p_ += 9 * 96 * 32;
  h16* c5b = p_; p_ += 9 * 32 * 32;

  // ---- pack weights into WMMA B-fragment layout ----
#define PD(src, dst, K, Kp, N) \
  pack_dense<<<(((Kp)/32)*((N)/16)*256 + 255)/256, 256, 0, stream>>>(src, dst, K, Kp, N)
#define PC(src, dst, O, I, Ip) \
  pack_conv<<<((9*(Ip)/32)*((O)/16)*256 + 255)/256, 256, 0, stream>>>(src, dst, O, I, Ip)
  PD(W0, w0p, 63, 64, 128);
  PD(W1, w1p, 128, 128, 128);
  PD(W2, w2p, 128, 128, 128);
  PD(W3, w3p, 128, 128, 128);
  PD(Wd, wdp, 155, 160, 128);
  PD(Wo, wop, 128, 128, 16);
  PC(w1a, c1a, 32, 16, 32);
  PC(w1b, c1b, 32, 32, 32);
  PC(w2a, c2a, 64, 32, 32);
  PC(w2b, c2b, 64, 64, 64);
  PC(w3a, c3a, 128, 64, 64);
  PC(w3b, c3b, 128, 128, 128);
  PC(w4a, c4a, 64, 192, 192);
  PC(w4b, c4b, 64, 64, 64);
  PC(w5a, c5a, 32, 96, 96);
  PC(w5b, c5b, 32, 32, 32);
#undef PD
#undef PC

  // ---- per-pixel MLP (WMMA, f16 act / f32 accum); writes NHWC + pad ----
  const size_t mlpLds = 8 * 2 * 16 * 160 * sizeof(h16); // 80 KB
  mlp_kernel<<<HWPIX / 128, 256, mlpLds, stream>>>(
      zbuf, ray, xyz, w0p, b0, w1p, b1, w2p, b2, w3p, b3, wdp, bd, wop, bo,
      FMAP, out + 2 * (size_t)HWPIX * 3);

  // ---- U-Net (implicit-GEMM WMMA convs, async double-buffered staging) ----
  conv3x3_kernel<2><<<HWPIX / 128, 256, 0, stream>>>(FMAP, c1a, b1a, T1, 32, 512, 512);
  conv3x3_kernel<2><<<HWPIX / 128, 256, 0, stream>>>(T1, c1b, b1b, E1, 32, 512, 512);
  pool2_kernel<<<(int)((32 * HW / 4 + 255) / 256), 256, 0, stream>>>(E1, P1, 32, 512, 512);
  conv3x3_kernel<4><<<HWPIX / 4 / 128, 256, 0, stream>>>(P1, c2a, b2a, T2, 32, 256, 256);
  conv3x3_kernel<4><<<HWPIX / 4 / 128, 256, 0, stream>>>(T2, c2b, b2b, E2, 64, 256, 256);
  pool2_kernel<<<(int)((64 * HW / 16 + 255) / 256), 256, 0, stream>>>(E2, P2, 64, 256, 256);
  conv3x3_kernel<8><<<HWPIX / 16 / 128, 256, 0, stream>>>(P2, c3a, b3a, T3, 64, 128, 128);
  conv3x3_kernel<8><<<HWPIX / 16 / 128, 256, 0, stream>>>(T3, c3b, b3b, BMu, 128, 128, 128);
  upcat_kernel<<<(int)((192 * HW / 4 + 255) / 256), 256, 0, stream>>>(BMu, E2, CAT2, 128, 64, 256, 256);
  conv3x3_kernel<4><<<HWPIX / 4 / 128, 256, 0, stream>>>(CAT2, c4a, b4a, T4, 192, 256, 256);
  conv3x3_kernel<4><<<HWPIX / 4 / 128, 256, 0, stream>>>(T4, c4b, b4b, U2, 64, 256, 256);
  upcat_kernel<<<(int)((96 * HW + 255) / 256), 256, 0, stream>>>(U2, E1, CAT1, 64, 32, 512, 512);
  conv3x3_kernel<2><<<HWPIX / 128, 256, 0, stream>>>(CAT1, c5a, b5a, T5, 96, 512, 512);
  conv3x3_kernel<2><<<HWPIX / 128, 256, 0, stream>>>(T5, c5b, b5b, U1, 32, 512, 512);

  // ---- 1x1 conv + sigmoid + mask composite + gt passthrough ----
  final_kernel<<<(HWPIX + 255) / 256, 256, 0, stream>>>(U1, wout, bout, zbuf, gt, out);
}
